// QGRULayer_60387240181942
// MI455X (gfx1250) — compile-verified, matrix operand order checked by reading, not run on hardware
//
#include <hip/hip_runtime.h>
#include <hip/hip_bf16.h>

// ---------------------------------------------------------------------------
// QGRU layer for MI455X (gfx1250): f16 WMMA GEMMs fed by async-to-LDS copies
// (ASYNCcnt double buffering) + chunked parallel scan.
// ---------------------------------------------------------------------------

typedef _Float16 half16v __attribute__((ext_vector_type(16)));
typedef _Float16 half8v  __attribute__((ext_vector_type(8)));
typedef float    float8v __attribute__((ext_vector_type(8)));
typedef float    float4v __attribute__((ext_vector_type(4)));
typedef int      int4v   __attribute__((ext_vector_type(4)));

#define B_DIM 8
#define L_DIM 4096
#define D_DIM 1024
#define M_DIM (B_DIM * L_DIM)          // 32768 rows when x viewed as [M, D]
#define KB 32                          // K step per WMMA (f16 16x16x32)
#define LDT 40                         // padded LDS row stride in halves (80 B)
#define NCHUNK 64                      // scan chunks along L
#define SCHUNK (L_DIM / NCHUNK)        // 64 steps per chunk
#define BLD ((size_t)B_DIM * (size_t)L_DIM * (size_t)D_DIM)
#define DD  ((size_t)D_DIM * (size_t)D_DIM)

__device__ __forceinline__ float8v zero8() {
    float8v z = {0.f, 0.f, 0.f, 0.f, 0.f, 0.f, 0.f, 0.f};
    return z;
}

__device__ __forceinline__ half8v cvt8(float4v a, float4v b) {
    half8v h;
    h[0] = (_Float16)a[0]; h[1] = (_Float16)a[1];
    h[2] = (_Float16)a[2]; h[3] = (_Float16)a[3];
    h[4] = (_Float16)b[0]; h[5] = (_Float16)b[1];
    h[6] = (_Float16)b[2]; h[7] = (_Float16)b[3];
    return h;
}

// ---- CDNA5 async global->LDS copy (16B per lane, tracked by ASYNCcnt) ----
__device__ __forceinline__ void async_b128(_Float16* lds_dst, const _Float16* gsrc) {
#if __has_builtin(__builtin_amdgcn_global_load_async_to_lds_b128)
    __builtin_amdgcn_global_load_async_to_lds_b128((int4v*)gsrc, (int4v*)lds_dst, 0, 0);
#else
    *(half8v*)lds_dst = *(const half8v*)gsrc;   // sync fallback
#endif
}

__device__ __forceinline__ void wait_async0() {
#if __has_builtin(__builtin_amdgcn_s_wait_asynccnt)
    __builtin_amdgcn_s_wait_asynccnt(0);
#else
    asm volatile("s_wait_asynccnt 0x0" ::: "memory");
#endif
}

// Load one 16x32 (f16) A/B fragment from a row-major LDS tile.
// ISA layout (CDNA5 ISA): lanes 0-15 = rows 0-15 holding K 0..7 and
// 16..23; lanes 16-31 = rows 0-15 holding K 8..15 and 24..31.  Two aligned
// ds_load_b128 per fragment; LDT=40 halves (80 B) avoids bank conflicts.
__device__ __forceinline__ half16v frag_ld(const _Float16* base, int row0, int lane) {
    const int r  = row0 + (lane & 15);
    const int ko = (lane & 16) ? 8 : 0;
    const _Float16* p = base + r * LDT + ko;
    half8v lo = *(const half8v*)(p);
    half8v hi = *(const half8v*)(p + 16);
    return __builtin_shufflevector(lo, hi, 0, 1, 2, 3, 4, 5, 6, 7,
                                           8, 9, 10, 11, 12, 13, 14, 15);
}

__device__ __forceinline__ float8v wmma16(half16v a, half16v b, float8v c) {
    // (neg_a, A, neg_b, B, c_mod, C, reuse_a, reuse_b)
    return __builtin_amdgcn_wmma_f32_16x16x32_f16(false, a, false, b,
                                                  (short)0, c, false, false);
}

// ---------------------------------------------------------------------------
// fp32 -> f16 conversion pass (8 elements / thread, fully coalesced).
// ---------------------------------------------------------------------------
__global__ __launch_bounds__(256) void cvt_kernel(const float* __restrict__ src,
                                                  _Float16* __restrict__ dst)
{
    const size_t i = ((size_t)blockIdx.x * 256 + threadIdx.x) * 8;
    float4v a = *(const float4v*)(src + i);
    float4v b = *(const float4v*)(src + i + 4);
    *(half8v*)(dst + i) = cvt8(a, b);
}

// ---------------------------------------------------------------------------
// Kernel 1: fused gate GEMMs (f16 inputs staged by async-to-LDS, double buffer).
//   z_f = x@W_f^T + b_f ; remember = sigmoid(z_f) * lin ; forget = 1-remember
//   z_i = x@W_i^T + b_i ; inp = tanh(z_i) * remember
// forget -> fbuf (y region of d_out, reused), inp -> ibuf (h region of d_out).
// Block tile 64(M) x 128(N), 8 waves each owning 32x32 per gate.
// ---------------------------------------------------------------------------
__global__ __launch_bounds__(256) void gates_kernel(
    const _Float16* __restrict__ xh,
    const _Float16* __restrict__ wfh, const float* __restrict__ bf,
    const _Float16* __restrict__ wih, const float* __restrict__ bi,
    float* __restrict__ fbuf, float* __restrict__ ibuf)
{
    __shared__ __align__(16) _Float16 sA[2][64 * LDT];
    __shared__ __align__(16) _Float16 sBf[2][128 * LDT];
    __shared__ __align__(16) _Float16 sBi[2][128 * LDT];

    const int m0   = blockIdx.x * 64;
    const int n0   = blockIdx.y * 128;
    const int tid  = threadIdx.x;
    const int lane = tid & 31;
    const int wave = tid >> 5;
    const int wm   = (wave & 1) * 32;   // wave row offset within block tile
    const int wn   = (wave >> 1) * 32;  // wave col offset within block tile

    float8v accf[2][2], acci[2][2];
#pragma unroll
    for (int i = 0; i < 2; i++)
#pragma unroll
        for (int j = 0; j < 2; j++) { accf[i][j] = zero8(); acci[i][j] = zero8(); }

    // Async loader roles: A tile 64x32 halves = 256 x 16B chunks (1/thread);
    // each W tile 128x32 halves = 512 chunks (2/thread).
    const int ar = tid >> 2, ac = (tid & 3) * 8;     // A: row, 8-half chunk
    const int wr = tid >> 1, wc = (tid & 1) * 16;    // W: row, 16-half chunk pair
    const _Float16* xrow = xh  + (size_t)(m0 + ar) * D_DIM + ac;
    const _Float16* frow = wfh + (size_t)(n0 + wr) * D_DIM + wc;
    const _Float16* irow = wih + (size_t)(n0 + wr) * D_DIM + wc;
    const int aoff = ar * LDT + ac;
    const int woff = wr * LDT + wc;

    // Prologue: fill buffer 0 with K-step 0.
    async_b128(&sA [0][aoff],     xrow);
    async_b128(&sBf[0][woff],     frow);
    async_b128(&sBf[0][woff + 8], frow + 8);
    async_b128(&sBi[0][woff],     irow);
    async_b128(&sBi[0][woff + 8], irow + 8);

    int kb = 0;
    for (int k0 = 0; k0 < D_DIM; k0 += KB, kb ^= 1) {
        wait_async0();        // own copies into buf[kb] done
        __syncthreads();      // everyone's copies done; prev compute on kb^1 done

        if (k0 + KB < D_DIM) {
            const int kn = k0 + KB;
            async_b128(&sA [kb ^ 1][aoff],     xrow + kn);
            async_b128(&sBf[kb ^ 1][woff],     frow + kn);
            async_b128(&sBf[kb ^ 1][woff + 8], frow + kn + 8);
            async_b128(&sBi[kb ^ 1][woff],     irow + kn);
            async_b128(&sBi[kb ^ 1][woff + 8], irow + kn + 8);
        }

        // ---- matrix core: 8 v_wmma per wave per K step ----
        half16v A0 = frag_ld(&sA[kb][0], wm, lane);
        half16v A1 = frag_ld(&sA[kb][0], wm + 16, lane);
#pragma unroll
        for (int cn = 0; cn < 2; cn++) {
            half16v Bf = frag_ld(&sBf[kb][0], wn + cn * 16, lane);
            half16v Bi = frag_ld(&sBi[kb][0], wn + cn * 16, lane);
            accf[0][cn] = wmma16(A0, Bf, accf[0][cn]);
            accf[1][cn] = wmma16(A1, Bf, accf[1][cn]);
            acci[0][cn] = wmma16(A0, Bi, acci[0][cn]);
            acci[1][cn] = wmma16(A1, Bi, acci[1][cn]);
        }
    }

    // ---- epilogue: gate nonlinearities, scatter per C/D lane layout ----
    const float invDm1 = 1.0f / (float)(D_DIM - 1);
#pragma unroll
    for (int rt = 0; rt < 2; rt++) {
#pragma unroll
        for (int cn = 0; cn < 2; cn++) {
            const int col  = n0 + wn + cn * 16 + (lane & 15);
            const int row0 = m0 + wm + rt * 16 + ((lane >> 4) << 3);
            const float linc = (float)col * invDm1;
            const float bff = bf[col];
            const float bii = bi[col];
#pragma unroll
            for (int v = 0; v < 8; v++) {
                const size_t idx = (size_t)(row0 + v) * D_DIM + col;
                const float zf  = accf[rt][cn][v] + bff;
                const float rem = linc / (1.0f + __expf(-zf));
                const float zi  = acci[rt][cn][v] + bii;
                fbuf[idx] = 1.0f - rem;
                ibuf[idx] = tanhf(zi) * rem;
            }
        }
    }
}

// ---------------------------------------------------------------------------
// Kernel 2a: in-place local scan per (b, d, chunk).
//   fbuf[l] <- cumprod of forget within chunk
//   ibuf[l] <- h_local (recurrence with zero initial state)
// ---------------------------------------------------------------------------
__global__ __launch_bounds__(256) void scan_local_kernel(
    float* __restrict__ fbuf, float* __restrict__ ibuf)
{
    const int d = blockIdx.x * 256 + threadIdx.x;
    const int c = blockIdx.y;
    const int b = blockIdx.z;
    size_t idx = ((size_t)b * L_DIM + (size_t)c * SCHUNK) * D_DIM + d;
    float cf = 1.0f, hl = 0.0f;
    for (int s = 0; s < SCHUNK; s++) {
        const float f = fbuf[idx];
        const float u = ibuf[idx];
        hl = f * hl + u;
        cf *= f;
        fbuf[idx] = cf;
        ibuf[idx] = hl;
        idx += D_DIM;
    }
}

// ---------------------------------------------------------------------------
// Kernel 2b: sequential scan over chunk summaries (chunk-end elements),
// seeded with the initial hidden state; stores incoming h per chunk to ws.
// ---------------------------------------------------------------------------
__global__ __launch_bounds__(256) void scan_chunk_kernel(
    const float* __restrict__ fbuf, const float* __restrict__ ibuf,
    const float* __restrict__ hidden, float* __restrict__ hin)
{
    const int d = blockIdx.x * 256 + threadIdx.x;
    const int b = blockIdx.y;
    float hcur = hidden[(size_t)b * D_DIM + d];
    for (int c = 0; c < NCHUNK; c++) {
        hin[((size_t)b * NCHUNK + c) * D_DIM + d] = hcur;
        const size_t idx =
            ((size_t)b * L_DIM + (size_t)c * SCHUNK + (SCHUNK - 1)) * D_DIM + d;
        hcur = ibuf[idx] + fbuf[idx] * hcur;   // U_c + A_c * h_in
    }
}

// ---------------------------------------------------------------------------
// Kernel 2c: h_t = h_local_t + cumprod_forget_t * h_in(chunk)   (in place)
// ---------------------------------------------------------------------------
__global__ __launch_bounds__(256) void scan_apply_kernel(
    const float* __restrict__ fbuf, float* __restrict__ hbuf,
    const float* __restrict__ hin)
{
    const int d = blockIdx.x * 256 + threadIdx.x;
    const int c = blockIdx.y;
    const int b = blockIdx.z;
    const float h0 = hin[((size_t)b * NCHUNK + c) * D_DIM + d];
    size_t idx = ((size_t)b * L_DIM + (size_t)c * SCHUNK) * D_DIM + d;
    for (int s = 0; s < SCHUNK; s++) {
        hbuf[idx] = hbuf[idx] + fbuf[idx] * h0;
        idx += D_DIM;
    }
}

// ---------------------------------------------------------------------------
// Kernel 3: y = h @ W_o^T + b_o (same async double-buffered WMMA skeleton).
// Overwrites the y region (which held cumprod_forget).
// ---------------------------------------------------------------------------
__global__ __launch_bounds__(256) void out_gemm_kernel(
    const _Float16* __restrict__ hh, const _Float16* __restrict__ woh,
    const float* __restrict__ bo, float* __restrict__ y)
{
    __shared__ __align__(16) _Float16 sA[2][64 * LDT];
    __shared__ __align__(16) _Float16 sB[2][128 * LDT];

    const int m0   = blockIdx.x * 64;
    const int n0   = blockIdx.y * 128;
    const int tid  = threadIdx.x;
    const int lane = tid & 31;
    const int wave = tid >> 5;
    const int wm   = (wave & 1) * 32;
    const int wn   = (wave >> 1) * 32;

    float8v acc[2][2];
#pragma unroll
    for (int i = 0; i < 2; i++)
#pragma unroll
        for (int j = 0; j < 2; j++) acc[i][j] = zero8();

    const int ar = tid >> 2, ac = (tid & 3) * 8;
    const int wr = tid >> 1, wc = (tid & 1) * 16;
    const _Float16* arow = hh  + (size_t)(m0 + ar) * D_DIM + ac;
    const _Float16* wrow = woh + (size_t)(n0 + wr) * D_DIM + wc;
    const int aoff = ar * LDT + ac;
    const int woff = wr * LDT + wc;

    async_b128(&sA[0][aoff],     arow);
    async_b128(&sB[0][woff],     wrow);
    async_b128(&sB[0][woff + 8], wrow + 8);

    int kb = 0;
    for (int k0 = 0; k0 < D_DIM; k0 += KB, kb ^= 1) {
        wait_async0();
        __syncthreads();

        if (k0 + KB < D_DIM) {
            const int kn = k0 + KB;
            async_b128(&sA[kb ^ 1][aoff],     arow + kn);
            async_b128(&sB[kb ^ 1][woff],     wrow + kn);
            async_b128(&sB[kb ^ 1][woff + 8], wrow + kn + 8);
        }

        half16v A0 = frag_ld(&sA[kb][0], wm, lane);
        half16v A1 = frag_ld(&sA[kb][0], wm + 16, lane);
#pragma unroll
        for (int cn = 0; cn < 2; cn++) {
            half16v Bc = frag_ld(&sB[kb][0], wn + cn * 16, lane);
            acc[0][cn] = wmma16(A0, Bc, acc[0][cn]);
            acc[1][cn] = wmma16(A1, Bc, acc[1][cn]);
        }
    }

#pragma unroll
    for (int rt = 0; rt < 2; rt++) {
#pragma unroll
        for (int cn = 0; cn < 2; cn++) {
            const int col  = n0 + wn + cn * 16 + (lane & 15);
            const int row0 = m0 + wm + rt * 16 + ((lane >> 4) << 3);
            const float bb = bo[col];
#pragma unroll
            for (int v = 0; v < 8; v++) {
                const size_t idx = (size_t)(row0 + v) * D_DIM + col;
                y[idx] = acc[rt][cn][v] + bb;
            }
        }
    }
}

// ---------------------------------------------------------------------------
extern "C" void kernel_launch(void* const* d_in, const int* in_sizes, int n_in,
                              void* d_out, int out_size, void* d_ws, size_t ws_size,
                              hipStream_t stream)
{
    (void)in_sizes; (void)n_in; (void)out_size; (void)ws_size;

    const float* x      = (const float*)d_in[0];
    const float* hidden = (const float*)d_in[1];
    const float* Wf     = (const float*)d_in[2];
    const float* bf     = (const float*)d_in[3];
    const float* Wi     = (const float*)d_in[4];
    const float* bi     = (const float*)d_in[5];
    const float* Wo     = (const float*)d_in[6];
    const float* bo     = (const float*)d_in[7];

    float* y = (float*)d_out;        // [B, L, D]  (scratch: forget / cumprod)
    float* h = y + BLD;              // [B, L, D]  (scratch: inp / h_local)

    // Workspace layout (f16 copies + chunk seeds)
    _Float16* xh  = (_Float16*)d_ws;     // [M, D]  x (later reused for h)
    _Float16* wfh = xh + BLD;            // [D, D]
    _Float16* wih = wfh + DD;            // [D, D]
    _Float16* woh = wih + DD;            // [D, D]
    float*    hin = (float*)(woh + DD);  // [B, NCHUNK, D]

    const dim3 gemm_grid(M_DIM / 64, D_DIM / 128);   // 512 x 8
    const dim3 scan_grid(D_DIM / 256, NCHUNK, B_DIM);
    const dim3 chunk_grid(D_DIM / 256, B_DIM);
    const int  cvt_x_blocks = (int)(BLD / 8 / 256);  // 16384
    const int  cvt_w_blocks = (int)(DD / 8 / 256);   // 512

    // f32 -> f16 staging copies
    cvt_kernel<<<cvt_x_blocks, 256, 0, stream>>>(x,  xh);
    cvt_kernel<<<cvt_w_blocks, 256, 0, stream>>>(Wf, wfh);
    cvt_kernel<<<cvt_w_blocks, 256, 0, stream>>>(Wi, wih);
    cvt_kernel<<<cvt_w_blocks, 256, 0, stream>>>(Wo, woh);

    gates_kernel<<<gemm_grid, 256, 0, stream>>>(xh, wfh, bf, wih, bi, y, h);
    scan_local_kernel<<<scan_grid, 256, 0, stream>>>(y, h);
    scan_chunk_kernel<<<chunk_grid, 256, 0, stream>>>(y, h, hidden, hin);
    scan_apply_kernel<<<scan_grid, 256, 0, stream>>>(y, h, hin);

    // h (fp32, in d_out) -> f16 (reuse xh buffer; x is no longer needed)
    cvt_kernel<<<cvt_x_blocks, 256, 0, stream>>>(h, xh);
    out_gemm_kernel<<<gemm_grid, 256, 0, stream>>>(xh, woh, bo, y);
}
